// SED_xLSTM_57629871177858
// MI455X (gfx1250) — compile-verified, matrix operand order checked by reading, not run on hardware
//
#include <hip/hip_runtime.h>
#include <hip/hip_bf16.h>
#include <math.h>

// ---------------------------------------------------------------------------
// MI455X (gfx1250, wave32) implementation of the SED_xLSTM forward pass.
// Dense projections -> v_wmma_f32_16x16x32_f16 with fp16 operands staged in
// LDS via GLOBAL_LOAD_ASYNC_TO_LDS_B128 (ASYNCcnt), double buffered.
// Everything else is wave32 VALU work.
// ---------------------------------------------------------------------------

typedef __attribute__((ext_vector_type(16))) _Float16 v16h;
typedef __attribute__((ext_vector_type(8)))  _Float16 v8h;
typedef __attribute__((ext_vector_type(8)))  float    v8f;

#define NB    128      // batch
#define NS    12       // tokens
#define NE    768      // embed dim
#define NHID  512      // HEADS*DH
#define NHEAD 4
#define NDH   128
#define NMLP  3072

#define ACT_NONE 0
#define ACT_GELU 1
#define ACT_SILU 2
#define ACT_LOGSIG 3

// CDNA5 async global->LDS copy (16B per lane), tracked by ASYNCcnt.
__device__ __forceinline__ void g2l_b128(unsigned lds_addr, const _Float16* g)
{
    asm volatile("global_load_async_to_lds_b128 %0, %1, off"
                 :: "v"(lds_addr), "v"(g) : "memory");
}
__device__ __forceinline__ void wait_async0()
{
    asm volatile("s_wait_asynccnt 0x0" ::: "memory");
}

// ------------------------------ WMMA GEMM ----------------------------------
// C[M,N] = act((A[M,K] * Bt[N,K]^T + bias) * scale) + res
// A, Bt fp16 row-major over K (lda = ldb = K). M%128==0, N%64==0, K%32==0.
#define TBM 128
#define TBN 64
#define TBK 32
#define TBKP 40   // padded LDS row stride (halves): conflict-free ds_load_b128

__global__ void __launch_bounds__(256)
k_gemm_wmma(const _Float16* __restrict__ A, const _Float16* __restrict__ Bt,
            float* __restrict__ Cf, _Float16* __restrict__ C16,
            const float* __restrict__ bias, const float* __restrict__ res,
            int M, int N, int K, int act, float scale)
{
    __shared__ _Float16 sA[2][TBM * TBKP];
    __shared__ _Float16 sB[2][TBN * TBKP];

    const int tid  = threadIdx.x;
    const int lane = tid & 31;
    const int wave = tid >> 5;
    const int wm   = (wave >> 1) * 32;   // wave tile origin in M (0/32/64/96)
    const int wn   = (wave & 1) * 32;    // wave tile origin in N (0/32)
    const int bm   = blockIdx.y * TBM;
    const int bn   = blockIdx.x * TBN;

    // async loader mapping: A tile 128x32 halves (32B/thread), B tile 64x32 (16B/thread)
    const int arow = tid >> 1, acol = (tid & 1) * 16;
    const int brow = tid >> 2, bcol = (tid & 3) * 8;
    const _Float16* gA = A  + (size_t)(bm + arow) * K + acol;
    const _Float16* gB = Bt + (size_t)(bn + brow) * K + bcol;
    const unsigned ldsA[2] = { (unsigned)(uintptr_t)&sA[0][arow * TBKP + acol],
                               (unsigned)(uintptr_t)&sA[1][arow * TBKP + acol] };
    const unsigned ldsB[2] = { (unsigned)(uintptr_t)&sB[0][brow * TBKP + bcol],
                               (unsigned)(uintptr_t)&sB[1][brow * TBKP + bcol] };

    v8f acc[2][2];
#pragma unroll
    for (int i = 0; i < 2; ++i)
#pragma unroll
        for (int j = 0; j < 2; ++j)
            acc[i][j] = v8f{0.f,0.f,0.f,0.f,0.f,0.f,0.f,0.f};

    const int nk = K / TBK;

    auto load_tile = [&](int bufsel, int kt) {
        const _Float16* pa = gA + (size_t)kt * TBK;
        const _Float16* pb = gB + (size_t)kt * TBK;
        g2l_b128(ldsA[bufsel], pa);
        g2l_b128(ldsA[bufsel] + 16, pa + 8);
        g2l_b128(ldsB[bufsel], pb);
    };

    // preload tile 0 through the async engine
    load_tile(0, 0);
    wait_async0();
    __syncthreads();

    int buf = 0;
    const int hsel = lane >> 4;          // 0: K 0-7/16-23, 1: K 8-15/24-31
    const int l15  = lane & 15;
    const int kb   = hsel * 8;

    for (int kt = 0; kt < nk; ++kt) {
        if (kt + 1 < nk) {
            if (kt + 2 < nk) {           // keep near caches warm 2 tiles ahead
                __builtin_prefetch(gA + (size_t)(kt + 2) * TBK, 0, 3);
                __builtin_prefetch(gB + (size_t)(kt + 2) * TBK, 0, 3);
            }
            load_tile(buf ^ 1, kt + 1);  // async copy overlaps the WMMAs below
        }

        union HV { v16h v; v8h h[2]; };
        HV a[2], b[2];
#pragma unroll
        for (int i = 0; i < 2; ++i) {
            const int mrow = wm + i * 16 + l15;
            a[i].h[0] = *(const v8h*)&sA[buf][mrow * TBKP + kb];
            a[i].h[1] = *(const v8h*)&sA[buf][mrow * TBKP + kb + 16];
            const int nrow = wn + i * 16 + l15;
            b[i].h[0] = *(const v8h*)&sB[buf][nrow * TBKP + kb];
            b[i].h[1] = *(const v8h*)&sB[buf][nrow * TBKP + kb + 16];
        }
#pragma unroll
        for (int i = 0; i < 2; ++i)
#pragma unroll
            for (int j = 0; j < 2; ++j)
                acc[i][j] = __builtin_amdgcn_wmma_f32_16x16x32_f16(
                    false, a[i].v, false, b[j].v, (short)0, acc[i][j], false, false);

        wait_async0();                   // this wave's async LDS writes landed
        __syncthreads();                 // ... and everyone else's
        buf ^= 1;
    }

    // epilogue (C/D layout: lane<16 -> M=r, lane>=16 -> M=r+8; N = lane&15)
    const int rofs = (lane >> 4) * 8;
#pragma unroll
    for (int i = 0; i < 2; ++i) {
#pragma unroll
        for (int j = 0; j < 2; ++j) {
            const int col = bn + wn + j * 16 + l15;
            const float bv = bias ? bias[col] : 0.f;
#pragma unroll
            for (int r = 0; r < 8; ++r) {
                const int row = bm + wm + i * 16 + rofs + r;
                float v = (acc[i][j][r] + bv) * scale;
                if (act == ACT_GELU)      v = 0.5f * v * (1.0f + erff(v * 0.70710678118f));
                else if (act == ACT_SILU) v = v / (1.0f + __expf(-v));
                const size_t o = (size_t)row * N + col;
                if (res) v += res[o];
                if (Cf)  Cf[o]  = v;
                if (C16) C16[o] = (_Float16)v;
            }
        }
    }
}

// ----------------------------- support kernels -----------------------------
__global__ void __launch_bounds__(256)
k_f16cvt(const float* __restrict__ x, _Float16* __restrict__ y, int n)
{
    int i = blockIdx.x * 256 + threadIdx.x;
    if (i < n) y[i] = (_Float16)x[i];
}

// W (K,N) f32 row-major -> Wt (N,K) f16 row-major
__global__ void __launch_bounds__(256)
k_transpose_f16(const float* __restrict__ W, _Float16* __restrict__ Wt, int K, int N)
{
    int i = blockIdx.x * 256 + threadIdx.x;
    if (i >= K * N) return;
    int n = i / K, k = i % K;
    Wt[i] = (_Float16)W[(size_t)k * N + n];
}

// rowwise layernorm; gmod>0 -> gamma/beta base = (row%gmod)*D (head-norm)
__global__ void __launch_bounds__(256)
k_ln_rows(const float* __restrict__ X, float* __restrict__ Yf,
          _Float16* __restrict__ Y16, const float* __restrict__ g,
          const float* __restrict__ bta, int D, int gmod, int act)
{
    const int row = blockIdx.x;
    const float* x = X + (size_t)row * D;
    __shared__ float r1[256], r2[256];
    float s = 0.f, s2 = 0.f;
    for (int i = threadIdx.x; i < D; i += 256) { float v = x[i]; s += v; s2 += v * v; }
    r1[threadIdx.x] = s; r2[threadIdx.x] = s2;
    __syncthreads();
    for (int o = 128; o > 0; o >>= 1) {
        if ((int)threadIdx.x < o) { r1[threadIdx.x] += r1[threadIdx.x + o]; r2[threadIdx.x] += r2[threadIdx.x + o]; }
        __syncthreads();
    }
    const float mean = r1[0] / D;
    const float var  = r2[0] / D - mean * mean;
    const float rstd = rsqrtf(var + 1e-5f);
    const size_t gofs = (gmod > 0) ? (size_t)(row % gmod) * D : 0;
    for (int i = threadIdx.x; i < D; i += 256) {
        float v = (x[i] - mean) * rstd;
        if (g)   v *= g[gofs + i];
        if (bta) v += bta[gofs + i];
        if (act == 1) v = fmaxf(v, 0.f);
        const size_t o = (size_t)row * D + i;
        if (Yf)  Yf[o]  = v;
        if (Y16) Y16[o] = (_Float16)v;
    }
}

// causal depthwise conv (kernel 4) over S=12 + SiLU; input = left half of UP
__global__ void __launch_bounds__(256)
k_cconv_silu(const float* __restrict__ UP, const float* __restrict__ w,
             const float* __restrict__ cb, float* __restrict__ CC,
             _Float16* __restrict__ CC16)
{
    int i = blockIdx.x * 256 + threadIdx.x;
    if (i >= NB * NS * NHID) return;
    const int ch = i % NHID, r = i / NHID, b = r / NS, s = r % NS;
    float acc = cb[ch];
#pragma unroll
    for (int t = 0; t < 4; ++t) {
        int ss = s - 3 + t;
        if (ss >= 0) acc += UP[(size_t)(b * NS + ss) * (2 * NHID) + ch] * w[ch * 4 + t];
    }
    float v = acc / (1.0f + __expf(-acc));
    CC[i] = v; CC16[i] = (_Float16)v;
}

// naive GEMM for skinny N (gates N=4, head N=4). W is (K,N) f32 row-major.
__global__ void __launch_bounds__(256)
k_gemm_naive(const float* __restrict__ A, const float* __restrict__ W,
             const float* __restrict__ bias, float* __restrict__ C,
             int M, int N, int K, int act)
{
    int i = blockIdx.x * 256 + threadIdx.x;
    if (i >= M * N) return;
    const int r = i / N, c = i % N;
    float acc = bias ? bias[c] : 0.f;
    for (int k = 0; k < K; ++k) acc += A[(size_t)r * K + k] * W[(size_t)k * N + c];
    if (act == ACT_LOGSIG) acc = fminf(acc, 0.f) - log1pf(__expf(-fabsf(acc)));
    C[i] = acc;
}

// exp-gated mLSTM recurrence: one workgroup per (b, head); C (128x128) in regs
__global__ void __launch_bounds__(256)
k_mlstm_rec(const float* __restrict__ Q, const float* __restrict__ Kk,
            const float* __restrict__ V, int vld,
            const float* __restrict__ IL, const float* __restrict__ FL,
            float* __restrict__ HO)
{
    const int blk = blockIdx.x, b = blk >> 2, h = blk & 3;
    const int t = threadIdx.x, d = t & 127, eh = t >> 7, e0 = eh * 64;
    __shared__ float qs[128], ks[128], vs[128], ns[128], red[128];
    __shared__ float hred[2][128];
    __shared__ float mS[1];
    float C[64];
#pragma unroll
    for (int j = 0; j < 64; ++j) C[j] = 0.f;
    if (t < 128) ns[t] = 0.f;
    if (t == 0) mS[0] = -1e30f;
    __syncthreads();

    for (int s = 0; s < NS; ++s) {
        const int rowq = (b * NS + s) * NHID + h * NDH;
        if (t < 128) { qs[t] = Q[rowq + t]; ks[t] = Kk[rowq + t]; }
        else { int tt = t - 128; vs[tt] = V[(size_t)(b * NS + s) * vld + h * NDH + tt]; }
        __syncthreads();

        const float il = IL[(b * NS + s) * NHEAD + h];
        const float fl = FL[(b * NS + s) * NHEAD + h];
        const float m  = mS[0];
        const float mn = fmaxf(fl + m, il);
        const float i_ = __expf(il - mn);
        const float f_ = __expf(fl + m - mn);

        const float vd = vs[d];
        float hp = 0.f;
#pragma unroll
        for (int j = 0; j < 64; ++j) {
            const float c = f_ * C[j] + i_ * vd * ks[e0 + j];
            C[j] = c;
            hp += c * qs[e0 + j];
        }
        hred[eh][d] = hp;
        if (t < 128) ns[t] = f_ * ns[t] + i_ * ks[t];
        __syncthreads();
        if (t < 128) red[t] = ns[t] * qs[t];
        __syncthreads();
        for (int o = 64; o > 0; o >>= 1) {
            if (t < o) red[t] += red[t + o];
            __syncthreads();
        }
        const float denom = fmaxf(fabsf(red[0]), 1.0f);
        if (t < 128) HO[rowq + t] = (hred[0][t] + hred[1][t]) / denom;
        if (t == 0) mS[0] = mn;
        __syncthreads();
    }
}

// GT16 = f16(HN * silu(right half of UP))
__global__ void __launch_bounds__(256)
k_gate_mul(const float* __restrict__ HN, const float* __restrict__ UP,
           _Float16* __restrict__ GT16)
{
    int i = blockIdx.x * 256 + threadIdx.x;
    if (i >= NB * NS * NHID) return;
    const int r = i / NHID, c = i % NHID;
    const float rv = UP[(size_t)r * (2 * NHID) + NHID + c];
    GT16[i] = (_Float16)(HN[i] * (rv / (1.0f + __expf(-rv))));
}

// generic conv1d over channels (NCH): y[b,o,l] = act(sum x[b,i,l+t-pad]*w[o,i,t]+b[o]) (+res)
__global__ void __launch_bounds__(256)
k_conv1d(const float* __restrict__ x, const float* __restrict__ w,
         const float* __restrict__ bias, const float* __restrict__ res,
         float* __restrict__ y, int Bn, int Cin, int Cout, int L, int KW,
         int pad, int act)
{
    int i = blockIdx.x * 256 + threadIdx.x;
    if (i >= Bn * Cout * L) return;
    const int l = i % L, o = (i / L) % Cout, b = i / (Cout * L);
    float acc = bias[o];
    for (int ci = 0; ci < Cin; ++ci)
        for (int t = 0; t < KW; ++t) {
            const int p = l - pad + t;
            if (p >= 0 && p < L)
                acc += x[((size_t)(b * Cin + ci)) * L + p] * w[(o * Cin + ci) * KW + t];
        }
    if (act == ACT_GELU) acc = 0.5f * acc * (1.0f + erff(acc * 0.70710678118f));
    if (res) acc += res[i];
    y[i] = acc;
}

// spatial phase: out[bp,o,h,w] = xspa[b,0..1,h,w] . pw[o,:] + pb[o]
__global__ void __launch_bounds__(256)
k_phase(const float* __restrict__ xsp, const float* __restrict__ pw,
        const float* __restrict__ pb, float* __restrict__ out, int bofs)
{
    int i = blockIdx.x * 256 + threadIdx.x;
    if (i >= 32 * 8 * NS * NE) return;
    const int w = i % NE; int t = i / NE;
    const int hh = t % NS; t /= NS;
    const int o = t % 8; const int bp = t / 8; const int b = bofs + bp;
    const float x0 = xsp[((size_t)(b * 2 + 0) * NS + hh) * NE + w];
    const float x1 = xsp[((size_t)(b * 2 + 1) * NS + hh) * NE + w];
    out[i] = x0 * pw[o * 2 + 0] + x1 * pw[o * 2 + 1] + pb[o];
}

__global__ void __launch_bounds__(256)
k_pool_meanmax(const float* __restrict__ H, float* __restrict__ P)
{
    int i = blockIdx.x * 256 + threadIdx.x;
    if (i >= 32 * NS * NE) return;
    const int w = i % NE; int t = i / NE;
    const int hh = t % NS; const int bp = t / NS;
    float s = 0.f, mx = -3.4e38f;
    for (int c = 0; c < 8; ++c) {
        const float v = H[((size_t)(bp * 8 + c) * NS + hh) * NE + w];
        s += v; mx = fmaxf(mx, v);
    }
    P[((size_t)(bp * 2 + 0) * NS + hh) * NE + w] = s * 0.125f;
    P[((size_t)(bp * 2 + 1) * NS + hh) * NE + w] = mx;
}

__global__ void __launch_bounds__(256)
k_conv3x3(const float* __restrict__ P, const float* __restrict__ saw,
          const float* __restrict__ sab, float* __restrict__ A)
{
    int i = blockIdx.x * 256 + threadIdx.x;
    if (i >= 32 * NS * NE) return;
    const int w = i % NE; int t = i / NE;
    const int hh = t % NS; const int bp = t / NS;
    float acc = sab[0];
    for (int c = 0; c < 2; ++c)
        for (int dy = -1; dy <= 1; ++dy)
            for (int dx = -1; dx <= 1; ++dx) {
                const int hy = hh + dy, wx = w + dx;
                if (hy >= 0 && hy < NS && wx >= 0 && wx < NE)
                    acc += P[((size_t)(bp * 2 + c) * NS + hy) * NE + wx] *
                           saw[(c * 3 + (dy + 1)) * 3 + (dx + 1)];
            }
    A[((size_t)bp * NS + hh) * NE + w] = acc;
}

__global__ void __launch_bounds__(256)
k_sa_gate(float* __restrict__ X, const float* __restrict__ H,
          const float* __restrict__ A)
{
    int i = blockIdx.x * 256 + threadIdx.x;
    if (i >= 32 * 8 * NS * NE) return;
    const int w = i % NE; int t = i / NE;
    const int hh = t % NS; t /= NS; t /= 8; const int bp = t;
    const float a = A[((size_t)bp * NS + hh) * NE + w];
    X[i] += tanhf(a) * H[i] + H[i];
}

// pack spa output (bp,c,p,w) chunk -> A2[(b*12+p), c*768+w] f16
__global__ void __launch_bounds__(256)
k_pack_spa(const float* __restrict__ S, _Float16* __restrict__ A2, int bofs)
{
    int i = blockIdx.x * 256 + threadIdx.x;
    if (i >= 32 * 8 * NS * NE) return;
    const int w = i % NE; int t = i / NE;
    const int p = t % NS; t /= NS;
    const int c = t % 8; const int bp = t / 8;
    const int row = (bofs + bp) * NS + p;
    A2[(size_t)row * (8 * NE) + c * NE + w] = (_Float16)S[i];
}

__global__ void __launch_bounds__(256)
k_add3pos(const float* __restrict__ T1, const float* __restrict__ T2,
          const float* __restrict__ pos, float* __restrict__ TOK)
{
    int i = blockIdx.x * 256 + threadIdx.x;
    if (i >= NB * NS * NE) return;
    const int col = i % NE; const int p = (i / NE) % NS;
    TOK[i] = T1[i] + T2[i] + pos[p * NE + col];
}

__global__ void __launch_bounds__(256)
k_scale2(float* __restrict__ X, int n)
{
    int i = blockIdx.x * 256 + threadIdx.x;
    if (i < n) X[i] *= 2.0f;
}

__global__ void __launch_bounds__(256)
k_silu_mul_add(const float* __restrict__ XC, const float* __restrict__ X1,
               const float* __restrict__ SK, float* __restrict__ Y, int n)
{
    int i = blockIdx.x * 256 + threadIdx.x;
    if (i >= n) return;
    const float v = XC[i];
    Y[i] = (v / (1.0f + __expf(-v))) * X1[i] + SK[i];
}

// ------------------------------- host side ---------------------------------
extern "C" void kernel_launch(void* const* d_in, const int* in_sizes, int n_in,
                              void* d_out, int out_size, void* d_ws, size_t ws_size,
                              hipStream_t stream)
{
    (void)in_sizes; (void)n_in; (void)out_size;
    const float* X_IN   = (const float*)d_in[0];   // (128,1,12,768)
    const float* XSPA   = (const float*)d_in[1];   // (128,2,12,768)
    auto F = [&](int i) { return (const float*)d_in[i]; };

    // ---- param leaf index map (jax sorted-key flatten) ----
    // dec[0]=2, dec[1]=46 ; within dec: conv1_b+0, conv1_w+1, conv2_b+2,
    // conv2_w+3, fc1_b+4, fc1_w+5, fc2_b+6, fc2_w+7, norm.b+8, norm.g+9,
    // xl0 + 10, xl1 + 27 (17 leaves per mlstm)
    // embed.b=90, embed.w=91 ; enc block i base 92+40i: fc1.b+0,fc1.w+1,
    // fc2.b+2,fc2.w+3,norm2.b+4,norm2.g+5,xl0+6,xl1+23
    // enc_norm.b=252,.g=253 ; head b1=254,b2=255,w1=256,w2=257 ; pos=258
    // spa blk base 259+10*blk: fc1.b+0,fc1.w+1,fc2.b+2,fc2.w+3,norm1.b+4,
    // norm1.g+5,norm2.b+6,norm2.g+7,sa_b+8,sa_w+9 ; spa.norm.b=279,.g=280 ;
    // phase.b=281,.w=282 ; spa_embed.b=283,.w=284
    // mlstm: conv_b+0,conv_w+1,down.b+2,down.w+3,fg.b+4,fg.w+5,hn_g+6,
    // ig.b+7,ig.w+8,k.b+9,k.w+10,ln.b+11,ln.g+12,q.b+13,q.w+14,up.b+15,up.w+16

    // ---- workspace ----
    char* ws = (char*)d_ws;
    size_t off = 0;
    auto alloc = [&](size_t elems, size_t esz) -> void* {
        void* p = ws + off;
        off += (elems * esz + 255) & ~(size_t)255;
        return p;
    };
    _Float16* WT16  = (_Float16*)alloc(9216ull * 3072, 2);   // transposed weight stage
    _Float16* A16   = (_Float16*)alloc(1536ull * 6144, 2);   // big fp16 A stage
    _Float16* H16   = (_Float16*)alloc(3072ull * 768, 2);    // LN fp16 out
    _Float16* HID16 = (_Float16*)alloc(3072ull * 3072, 2);   // MLP hidden fp16
    _Float16* CC16  = (_Float16*)alloc(1536ull * 512, 2);
    _Float16* GT16  = (_Float16*)alloc(1536ull * 512, 2);
    float* SX   = (float*)alloc(3072ull * 768, 4);
    float* SH   = (float*)alloc(3072ull * 768, 4);
    float* SP   = (float*)alloc(32ull * 2 * NS * NE, 4);
    float* SA   = (float*)alloc(32ull * NS * NE, 4);
    float* T1   = (float*)alloc(1536ull * 768, 4);
    float* T2   = (float*)alloc(1536ull * 768, 4);
    float* TOK  = (float*)alloc(1536ull * 768, 4);
    float* UP   = (float*)alloc(1536ull * 1024, 4);
    float* CC   = (float*)alloc(1536ull * 512, 4);
    float* QQ   = (float*)alloc(1536ull * 512, 4);
    float* KK   = (float*)alloc(1536ull * 512, 4);
    float* GI   = (float*)alloc(1536ull * 4, 4);
    float* GF   = (float*)alloc(1536ull * 4, 4);
    float* HO   = (float*)alloc(1536ull * 512, 4);
    float* HN   = (float*)alloc(1536ull * 512, 4);
    float* SKIP = (float*)alloc(1536ull * 768, 4);
    float* XC   = (float*)alloc(1536ull * 768, 4);
    float* X1   = (float*)alloc(1536ull * 768, 4);
    float* DCH  = (float*)alloc(128ull * 48 * 768, 4);
    float* DCH2 = (float*)alloc(128ull * 48 * 768, 4);
    if (off > ws_size) return;   // not enough scratch; bail deterministically

    auto g1d = [](size_t n) { return dim3((unsigned)((n + 255) / 256)); };

    auto ln = [&](const float* Xp, float* Yf, _Float16* Y16, const float* g,
                  const float* b, int rows, int D, int gmod, int act) {
        k_ln_rows<<<rows, 256, 0, stream>>>(Xp, Yf, Y16, g, b, D, gmod, act);
    };
    auto transw = [&](const float* W, int K, int N) {
        k_transpose_f16<<<g1d((size_t)K * N), 256, 0, stream>>>(W, WT16, K, N);
    };
    auto cvt = [&](const float* x, _Float16* y, size_t n) {
        k_f16cvt<<<g1d(n), 256, 0, stream>>>(x, y, (int)n);
    };
    auto gemm = [&](const _Float16* A, const _Float16* Bt, float* Cf,
                    _Float16* C16, const float* bias, const float* res,
                    int M, int N, int K, int act, float scale) {
        dim3 grid(N / TBN, M / TBM);
        k_gemm_wmma<<<grid, 256, 0, stream>>>(A, Bt, Cf, C16, bias, res, M, N, K, act, scale);
    };

    // ---- mLSTM layer: X (1536,768) updated in place (residual fused) ----
    auto run_mlstm = [&](int pb, float* Xp) {
        ln(Xp, nullptr, H16, F(pb + 12), F(pb + 11), 1536, NE, 0, 0);
        transw(F(pb + 16), NE, 2 * NHID);                                   // up.w
        gemm(H16, WT16, UP, nullptr, F(pb + 15), nullptr, 1536, 2 * NHID, NE, ACT_NONE, 1.f);
        k_cconv_silu<<<g1d(1536ull * NHID), 256, 0, stream>>>(UP, F(pb + 1), F(pb + 0), CC, CC16);
        transw(F(pb + 14), NHID, NHID);                                     // q.w
        gemm(CC16, WT16, QQ, nullptr, F(pb + 13), nullptr, 1536, NHID, NHID, ACT_NONE, 1.f);
        transw(F(pb + 10), NHID, NHID);                                     // k.w
        gemm(CC16, WT16, KK, nullptr, F(pb + 9), nullptr, 1536, NHID, NHID, ACT_NONE,
             0.08838834764831845f);                                         // 1/sqrt(128)
        k_gemm_naive<<<g1d(1536ull * NHEAD), 256, 0, stream>>>(CC, F(pb + 8), F(pb + 7), GI,
                                                               1536, NHEAD, NHID, ACT_NONE);
        k_gemm_naive<<<g1d(1536ull * NHEAD), 256, 0, stream>>>(CC, F(pb + 5), F(pb + 4), GF,
                                                               1536, NHEAD, NHID, ACT_LOGSIG);
        k_mlstm_rec<<<NB * NHEAD, 256, 0, stream>>>(QQ, KK, UP, 2 * NHID, GI, GF, HO);
        ln(HO, HN, nullptr, F(pb + 6), nullptr, 1536 * NHEAD, NDH, NHEAD, 0); // head norm * hn_g
        k_gate_mul<<<g1d(1536ull * NHID), 256, 0, stream>>>(HN, UP, GT16);
        transw(F(pb + 3), NHID, NE);                                        // down.w
        gemm(GT16, WT16, Xp, nullptr, F(pb + 2), Xp, 1536, NE, NHID, ACT_NONE, 1.f);
    };

    // ---- MLP: X += fc2(gelu(fc1(LN(X)))) ----
    auto run_mlp = [&](float* Xp, const float* ng, const float* nb,
                       const float* w1, const float* b1, const float* w2,
                       const float* b2, int rows) {
        ln(Xp, nullptr, H16, ng, nb, rows, NE, 0, 0);
        transw(w1, NE, NMLP);
        gemm(H16, WT16, nullptr, HID16, b1, nullptr, rows, NMLP, NE, ACT_GELU, 1.f);
        transw(w2, NMLP, NE);
        gemm(HID16, WT16, Xp, nullptr, b2, Xp, rows, NE, NMLP, ACT_NONE, 1.f);
    };

    // =============== spatial branch (4 batch chunks of 32) ================
    const size_t spaChunk = 32ull * 8 * NS * NE;
    for (int cb = 0; cb < 4; ++cb) {
        const int bofs = cb * 32;
        k_phase<<<g1d(spaChunk), 256, 0, stream>>>(XSPA, F(282), F(281), SX, bofs);
        ln(SX, SX, nullptr, F(280), F(279), 3072, NE, 0, 1);               // relu(LN)
        for (int blk = 0; blk < 2; ++blk) {
            const int sb = 259 + 10 * blk;
            ln(SX, SH, nullptr, F(sb + 5), F(sb + 4), 3072, NE, 0, 0);     // norm1
            k_pool_meanmax<<<g1d(32ull * NS * NE), 256, 0, stream>>>(SH, SP);
            k_conv3x3<<<g1d(32ull * NS * NE), 256, 0, stream>>>(SP, F(sb + 9), F(sb + 8), SA);
            k_sa_gate<<<g1d(spaChunk), 256, 0, stream>>>(SX, SH, SA);
            run_mlp(SX, F(sb + 7), F(sb + 6), F(sb + 1), F(sb + 0),
                    F(sb + 3), F(sb + 2), 3072);
        }
        ln(SX, SH, nullptr, F(280), F(279), 3072, NE, 0, 0);               // final spa LN
        k_pack_spa<<<g1d(spaChunk), 256, 0, stream>>>(SH, A16, bofs);
    }
    // t2 = packed_spa (1536 x 6144) @ spa_embed.w^T
    cvt(F(284), WT16, 768ull * 6144);                                      // already (N,K)
    gemm(A16, WT16, T2, nullptr, F(283), nullptr, 1536, NE, 6144, ACT_NONE, 1.f);

    // t1 = x (1536 x 768) @ embed.w^T
    cvt(X_IN, H16, 1536ull * 768);
    cvt(F(91), WT16, 768ull * 768);                                        // already (N,K)
    gemm(H16, WT16, T1, nullptr, F(90), nullptr, 1536, NE, NE, ACT_NONE, 1.f);
    k_add3pos<<<g1d(1536ull * 768), 256, 0, stream>>>(T1, T2, F(258), TOK);

    // ========================== encoder blocks ============================
    for (int i = 0; i < 4; ++i) {
        const int eb = 92 + 40 * i;
        run_mlstm(eb + 6, TOK);
        run_mlstm(eb + 23, TOK);
        k_scale2<<<g1d(1536ull * 768), 256, 0, stream>>>(TOK, 1536 * 768); // x = x + x
        run_mlp(TOK, F(eb + 5), F(eb + 4), F(eb + 1), F(eb + 0), F(eb + 3), F(eb + 2), 1536);
    }
    ln(TOK, TOK, nullptr, F(253), F(252), 1536, NE, 0, 0);                 // enc_norm

    // ============================ decoders ================================
    for (int di = 0; di < 2; ++di) {
        const int db = 2 + 44 * di;
        ln(TOK, TOK, nullptr, F(db + 9), F(db + 8), 1536, NE, 0, 0);
        hipMemcpyAsync(SKIP, TOK, 1536ull * 768 * 4, hipMemcpyDeviceToDevice, stream);
        k_conv1d<<<g1d(128ull * 12 * 768), 256, 0, stream>>>(TOK, F(db + 5), F(db + 4),
            nullptr, XC, NB, 12, 12, NE, 3, 1, ACT_NONE);                  // fc1 conv
        k_conv1d<<<g1d(128ull * 12 * 768), 256, 0, stream>>>(XC, F(db + 7), F(db + 6),
            nullptr, X1, NB, 12, 12, NE, 1, 0, ACT_NONE);                  // fc2 conv
        run_mlstm(db + 10, X1);
        run_mlstm(db + 27, X1);
        k_silu_mul_add<<<g1d(1536ull * 768), 256, 0, stream>>>(XC, X1, SKIP, TOK, 1536 * 768);
        k_conv1d<<<g1d(128ull * 48 * 768), 256, 0, stream>>>(TOK, F(db + 1), F(db + 0),
            nullptr, DCH, NB, 12, 48, NE, 3, 1, ACT_GELU);                 // conv1 + gelu
        ln(DCH, DCH2, nullptr, F(db + 9), F(db + 8), 128 * 48, NE, 0, 0);
        k_conv1d<<<g1d(128ull * 12 * 768), 256, 0, stream>>>(DCH2, F(db + 3), F(db + 2),
            SKIP, TOK, NB, 48, 12, NE, 3, 1, ACT_NONE);                    // conv2 + skip
    }

    // ============================== head ==================================
    cvt(TOK, A16, 1536ull * 768);                                          // view (128, 9216)
    transw(F(256), NS * NE, 4 * NE);                                       // w1 (9216,3072)
    gemm(A16, WT16, UP, nullptr, F(254), nullptr, NB, 4 * NE, NS * NE, ACT_GELU, 1.f);
    k_gemm_naive<<<g1d((size_t)NB * 4), 256, 0, stream>>>(UP, F(257), F(255),
        (float*)d_out, NB, 4, 4 * NE, ACT_NONE);
}
